// SelfAttention_12532714570093
// MI455X (gfx1250) — matrix operand order repaired and checked by reading, not compiled, not run on hardware
//
#include <hip/hip_runtime.h>
#include <hip/hip_bf16.h>

// ---------------------------------------------------------------------------
// MHSA for MI455X (gfx1250): all GEMMs via v_wmma_f32_16x16x32_f16 (wave32),
// V-tile staging via Tensor Data Mover (tensor_load_to_lds + s_wait_tensorcnt).
//   B=2, S=2048, D=1024, H=16, Dh=64
//   d_out = out[B*S*D] fp32  ||  attn[B*H*S*S] fp32
// ---------------------------------------------------------------------------

#define B_  2
#define S_  2048
#define D_  1024
#define H_  16
#define DH_ 64
#define MTOT_ (B_ * S_)                      // 4096
#define OUT_ELEMS_ ((size_t)B_ * S_ * D_)    // 4,194,304

typedef __attribute__((ext_vector_type(16))) _Float16 v16h;
typedef __attribute__((ext_vector_type(8)))  _Float16 v8h;
typedef __attribute__((ext_vector_type(4)))  _Float16 v4h;
typedef __attribute__((ext_vector_type(8)))  float    v8f;
typedef __attribute__((ext_vector_type(4)))  unsigned int u32x4;
typedef __attribute__((ext_vector_type(8)))  int      i32x8;
typedef __attribute__((ext_vector_type(4)))  int      i32x4;

__device__ __forceinline__ v8f wmma_f16(v16h a, v16h b, v8f c) {
    return __builtin_amdgcn_wmma_f32_16x16x32_f16(
        /*neg_a=*/false, a, /*neg_b=*/false, b,
        /*c_mod=*/(short)0, c, /*reuse_a=*/false, /*reuse_b=*/false);
}

// ---------------------------------------------------------------------------
// Tiled GEMM: C[M,N] = A[M,K] @ W[N,K]^T + bias   (nn.Linear semantics)
//   A: fp32 or f16 (TIN), W/bias fp32, out fp32 (d_out) or f16 (workspace).
//   Block tile 128x128x32, 256 threads (8 waves), wave tile 32x64.
// ---------------------------------------------------------------------------
#define GBM 128
#define GBN 128
#define GBK 32
#define LDT 48   // LDS row stride in f16 elems (96B, 16B-aligned, conflict-spread)

template <typename TIN, bool OUT_F32>
__global__ __launch_bounds__(256)
void gemm_bias_wmma(const TIN* __restrict__ A, const float* __restrict__ W,
                    const float* __restrict__ bias, void* __restrict__ Cout,
                    int M, int N, int K) {
    __shared__ _Float16 sA[GBM * LDT];
    __shared__ _Float16 sW[GBN * LDT];

    const int tid  = threadIdx.x;
    const int lane = tid & 31;
    const int wave = tid >> 5;
    const int wm   = wave >> 1;       // 0..3  (M direction)
    const int wn   = wave & 1;        // 0..1  (N direction)
    const int l15  = lane & 15;
    const int lhf  = lane >> 4;

    const int rowBase = blockIdx.y * GBM;
    const int colBase = blockIdx.x * GBN;

    const int lrow = tid >> 3;        // 0..31
    const int lcol = tid & 7;         // 0..7 (x4 elems)

    v8f acc[2][4] = {};

    for (int kt = 0; kt < K; kt += GBK) {
#pragma unroll
        for (int r = 0; r < 4; ++r) {
            const int row = lrow + 32 * r;
            // A tile -> sA[row][k] (convert to f16 if fp32 input)
            if constexpr (sizeof(TIN) == 4) {
                const float4 f = *(const float4*)((const float*)A +
                        (size_t)(rowBase + row) * K + kt + lcol * 4);
                v4h hA = { (_Float16)f.x, (_Float16)f.y, (_Float16)f.z, (_Float16)f.w };
                *(v4h*)&sA[row * LDT + lcol * 4] = hA;
            } else {
                const v4h hA = *(const v4h*)((const _Float16*)A +
                        (size_t)(rowBase + row) * K + kt + lcol * 4);
                *(v4h*)&sA[row * LDT + lcol * 4] = hA;
            }
            // W tile -> sW[n][k]
            const float4 g = *(const float4*)(W +
                    (size_t)(colBase + row) * K + kt + lcol * 4);
            v4h hW = { (_Float16)g.x, (_Float16)g.y, (_Float16)g.z, (_Float16)g.w };
            *(v4h*)&sW[row * LDT + lcol * 4] = hW;
        }
        __syncthreads();

        // A fragments (contiguous k = 16*lhf + s, same order as B)
        v16h af[2];
#pragma unroll
        for (int mt = 0; mt < 2; ++mt) {
            const int m = wm * 32 + mt * 16 + l15;
            const v8h lo = *(const v8h*)&sA[m * LDT + lhf * 16];
            const v8h hi = *(const v8h*)&sA[m * LDT + lhf * 16 + 8];
#pragma unroll
            for (int i = 0; i < 8; ++i) { af[mt][i] = lo[i]; af[mt][i + 8] = hi[i]; }
        }
        // B fragments: lane = column n, 16 contiguous k per lane-half
        v16h bf[4];
#pragma unroll
        for (int nt = 0; nt < 4; ++nt) {
            const int n = wn * 64 + nt * 16 + l15;
            bf[nt] = *(const v16h*)&sW[n * LDT + lhf * 16];
        }
#pragma unroll
        for (int mt = 0; mt < 2; ++mt)
#pragma unroll
            for (int nt = 0; nt < 4; ++nt)
                acc[mt][nt] = wmma_f16(af[mt], bf[nt], acc[mt][nt]);
        __syncthreads();
    }

    // Epilogue: bias add + store. C layout: m = r + 8*(lane>=16), n = lane&15.
#pragma unroll
    for (int nt = 0; nt < 4; ++nt) {
        const int ng = colBase + wn * 64 + nt * 16 + l15;
        const float bv = bias[ng];
#pragma unroll
        for (int mt = 0; mt < 2; ++mt) {
#pragma unroll
            for (int r = 0; r < 8; ++r) {
                const int mg = rowBase + wm * 32 + mt * 16 + r + 8 * lhf;
                const float val = acc[mt][nt][r] + bv;
                if constexpr (OUT_F32)
                    ((float*)Cout)[(size_t)mg * N + ng] = val;
                else
                    ((_Float16*)Cout)[(size_t)mg * N + ng] = (_Float16)val;
            }
        }
    }
}

// ---------------------------------------------------------------------------
// Fused attention: one workgroup per (b, h, 16-row query block).
//  Phase 1: scores = Q K^T / 8 -> LDS stripe [16][2052] fp32 (WMMA)
//  Phase 2: mask -> -1e9
//  Phase 3: softmax in LDS; write attn once to d_out; probs -> f16 Pbuf
//  Phase 4: ctx = P @ V; V tile DMA'd by the Tensor Data Mover into LDS,
//           transposed in LDS for B-fragments, WMMA accumulate, cross-wave
//           k-reduction, ctx f16 -> ws in [B,S,H*Dh] order (transpose free)
// ---------------------------------------------------------------------------
#define QR   16
#define SST  2052               // score row stride (fp32 elems)
#define VTS  72                 // Vt row stride (f16 elems, 144B, 16B-aligned)

#define SC_BYTES  (QR * SST * 4)         // 131328
#define PB_BYTES  (QR * S_ * 2)          // 65536
#define VT_BYTES  (DH_ * VTS * 2)        // 9216
#define RR_BYTES  (QR * 16 * 4)          // 1024
#define ST_BYTES  (32 * 4)               // 128
#define RB_BYTES  (8 * 256 * 4)          // 8192
#define VROW_OFF  (SC_BYTES + PB_BYTES + VT_BYTES + RR_BYTES + ST_BYTES + RB_BYTES)
#define VROW_BYTES (64 * DH_ * 2)        // 8192 (row-major TDM landing pad)
#define ATTN_SMEM (VROW_OFF + VROW_BYTES)

__global__ __launch_bounds__(256)
void attn_kernel(const _Float16* __restrict__ Qh, const _Float16* __restrict__ Kh,
                 const _Float16* __restrict__ Vh, const int* __restrict__ mask,
                 float* __restrict__ attnOut, _Float16* __restrict__ Ctxh) {
    extern __shared__ char smem[];
    float*    scores = (float*)smem;
    _Float16* Pbuf   = (_Float16*)(smem + SC_BYTES);
    _Float16* Vt     = (_Float16*)(smem + SC_BYTES + PB_BYTES);
    float*    rowred = (float*)(smem + SC_BYTES + PB_BYTES + VT_BYTES);
    float*    rowst  = (float*)(smem + SC_BYTES + PB_BYTES + VT_BYTES + RR_BYTES);
    float*    redbuf = (float*)(smem + SC_BYTES + PB_BYTES + VT_BYTES + RR_BYTES + ST_BYTES);
    _Float16* Vrow   = (_Float16*)(smem + VROW_OFF);

    const int tid  = threadIdx.x;
    const int lane = tid & 31;
    const int wave = tid >> 5;
    const int l15  = lane & 15;
    const int lhf  = lane >> 4;

    const int qblk = blockIdx.x & 127;         // S/QR = 128
    const int bh   = blockIdx.x >> 7;
    const int h    = bh & (H_ - 1);
    const int b    = bh >> 4;
    const int qbase = qblk * QR;

    // warm L2/WGP$ for the mask stripe we will read in phase 2
    __builtin_prefetch((const void*)(mask + ((size_t)b * S_ + qbase) * S_ + tid * 128), 0, 0);

    // ---- Phase 1: scores = Q K^T * (1/sqrt(64)) --------------------------
    const _Float16* qrow = Qh + ((size_t)b * S_ + qbase + l15) * D_ + h * DH_;
    v16h aQ[2];
#pragma unroll
    for (int ks = 0; ks < 2; ++ks) {
        const _Float16* p = qrow + ks * 32 + lhf * 16;
        const v8h lo = *(const v8h*)p;
        const v8h hi = *(const v8h*)(p + 8);
#pragma unroll
        for (int i = 0; i < 8; ++i) { aQ[ks][i] = lo[i]; aQ[ks][i + 8] = hi[i]; }
    }
    const int nstripe = wave * 256;            // each wave: 256 key columns
#pragma unroll 4
    for (int nt = 0; nt < 16; ++nt) {
        const int sk = nstripe + nt * 16 + l15;   // key position (= column n)
        v8f acc = {};
#pragma unroll
        for (int ks = 0; ks < 2; ++ks) {
            const v16h bK = *(const v16h*)(Kh + ((size_t)b * S_ + sk) * D_ +
                                           h * DH_ + ks * 32 + lhf * 16);
            acc = wmma_f16(aQ[ks], bK, acc);
        }
#pragma unroll
        for (int r = 0; r < 8; ++r)
            scores[(r + 8 * lhf) * SST + sk] = acc[r] * 0.125f;
    }
    __syncthreads();

    // ---- Phase 2: mask ---------------------------------------------------
    for (int i = tid; i < QR * S_; i += 256) {
        const int m = i >> 11, n = i & (S_ - 1);
        if (mask[((size_t)b * S_ + qbase + m) * S_ + n] == 0)
            scores[m * SST + n] = -1e9f;
    }
    __syncthreads();

    // ---- Phase 3: softmax (16 threads per row, 128 elems each) -----------
    const int srow = tid >> 4;         // 0..15
    const int part = tid & 15;
    const int n0   = part * 128;
    float mx = -3.4e38f;
    for (int i = 0; i < 128; ++i) mx = fmaxf(mx, scores[srow * SST + n0 + i]);
    rowred[srow * 16 + part] = mx;
    __syncthreads();
    if (tid < QR) {
        float m2 = rowred[tid * 16];
        for (int i = 1; i < 16; ++i) m2 = fmaxf(m2, rowred[tid * 16 + i]);
        rowst[tid] = m2;
    }
    __syncthreads();
    const float rmax = rowst[srow];
    float psum = 0.f;
    for (int i = 0; i < 128; ++i) {
        const float e = __expf(scores[srow * SST + n0 + i] - rmax);
        scores[srow * SST + n0 + i] = e;
        psum += e;
    }
    rowred[srow * 16 + part] = psum;
    __syncthreads();
    if (tid < QR) {
        float s2 = 0.f;
        for (int i = 0; i < 16; ++i) s2 += rowred[tid * 16 + i];
        rowst[16 + tid] = 1.0f / s2;
    }
    __syncthreads();
    const float rinv = rowst[16 + srow];
    float* arow = attnOut + (((size_t)(b * H_ + h)) * S_ + qbase + srow) * S_;
    for (int i = 0; i < 128; ++i) {
        const float p = scores[srow * SST + n0 + i] * rinv;
        arow[n0 + i] = p;                       // mandatory attn output (once)
        Pbuf[srow * S_ + n0 + i] = (_Float16)p;
    }
    __syncthreads();

    // ---- Phase 4: ctx = P @ V  (waves: 4 d-tiles x 2 k-halves) -----------
    const int dtile = wave & 3;
    const int ksel  = (wave >> 2) * 32;
    const unsigned ldsVrow = (unsigned)__builtin_amdgcn_groupstaticsize() + VROW_OFF;
    v8f accPV = {};
    for (int it = 0; it < S_ / 64; ++it) {
        const int kwin = it * 64;

        // (a) TDM: DMA V[kwin..kwin+63][0..63] (row-major f16) into LDS Vrow.
        //     2D tile: tile_dim0=64 (d), tile_dim1=64 (rows), row stride D_.
        //     Issued by wave 0 only (per-wave DMA engine, EXEC-independent).
        if (wave == 0) {
            const unsigned long long ga = (unsigned long long)(uintptr_t)
                (Vh + ((size_t)b * S_ + kwin) * D_ + h * DH_);
            u32x4 g0;
            g0[0] = 1u;                                        // count=1, no gather
            g0[1] = ldsVrow;                                   // lds_addr
            g0[2] = (unsigned)(ga & 0xFFFFFFFFu);              // global_addr[31:0]
            g0[3] = (unsigned)((ga >> 32) & 0x1FFFFFFu) | (2u << 30); // [56:32] | type=2
            i32x8 g1;
            g1[0] = (1 << 16);                                 // wg_mask=0, data_size=2B
            g1[1] = (int)((unsigned)(D_ & 0xFFFF) << 16);      // tensor_dim0[15:0]
            g1[2] = (int)(((unsigned)D_ >> 16) | ((unsigned)(S_ & 0xFFFF) << 16)); // dim0 hi | dim1 lo
            g1[3] = (int)(((unsigned)S_ >> 16) | (64u << 16)); // dim1 hi | tile_dim0=64
            g1[4] = (int)64;                                   // tile_dim1=64, tile_dim2=0
            g1[5] = (int)D_;                                   // tensor_dim0_stride (lo32)
            g1[6] = 0;
            g1[7] = 0;
            const i32x4 gz4 = {0, 0, 0, 0};
            const i32x8 gz8 = {0, 0, 0, 0, 0, 0, 0, 0};
            __builtin_amdgcn_tensor_load_to_lds(g0, g1, gz4, gz4, gz8, 0);
            __builtin_amdgcn_s_wait_tensorcnt(0);
        }
        __syncthreads();

        // (b) LDS->LDS transpose stage: Vt[d][k] for contiguous B-fragments
        {
            const int vk = tid >> 2;            // 0..63
            const int ch = tid & 3;             // 16 d's each
            const v16h vv = *(const v16h*)&Vrow[vk * DH_ + ch * 16];
#pragma unroll
            for (int i = 0; i < 16; ++i)
                Vt[(ch * 16 + i) * VTS + vk] = vv[i];
        }
        __syncthreads();

        // (c) WMMA: A frag from Pbuf (k-window = kwin + ksel), B frag from Vt
        v16h aP;
        {
            const _Float16* p = Pbuf + (size_t)l15 * S_ + kwin + ksel + lhf * 16;
            const v8h lo = *(const v8h*)p;
            const v8h hi = *(const v8h*)(p + 8);
#pragma unroll
            for (int i = 0; i < 8; ++i) { aP[i] = lo[i]; aP[i + 8] = hi[i]; }
        }
        const v16h bV = *(const v16h*)&Vt[(dtile * 16 + l15) * VTS + ksel + lhf * 16];
        accPV = wmma_f16(aP, bV, accPV);
        __syncthreads();
    }

    // cross-wave (k-half) reduction, then write ctx as [B,S,H*Dh] f16
#pragma unroll
    for (int r = 0; r < 8; ++r)
        redbuf[wave * 256 + r * 32 + lane] = accPV[r];
    __syncthreads();
    if (tid < 128) {
        const int w = tid >> 5;                 // = d-tile
        const int l = tid & 31;
#pragma unroll
        for (int r = 0; r < 8; ++r) {
            const float s = redbuf[w * 256 + r * 32 + l] +
                            redbuf[(w + 4) * 256 + r * 32 + l];
            const int m = qbase + r + 8 * (l >> 4);
            const int d = w * 16 + (l & 15);
            Ctxh[((size_t)b * S_ + m) * D_ + h * DH_ + d] = (_Float16)s;
        }
    }
}

// ---------------------------------------------------------------------------
extern "C" void kernel_launch(void* const* d_in, const int* in_sizes, int n_in,
                              void* d_out, int out_size, void* d_ws, size_t ws_size,
                              hipStream_t stream) {
    const float* queries = (const float*)d_in[0];
    const float* keys    = (const float*)d_in[1];
    const float* values  = (const float*)d_in[2];
    const int*   mask    = (const int*)d_in[3];
    const float* Wq = (const float*)d_in[4];  const float* bq = (const float*)d_in[5];
    const float* Wk = (const float*)d_in[6];  const float* bk = (const float*)d_in[7];
    const float* Wv = (const float*)d_in[8];  const float* bv = (const float*)d_in[9];
    const float* Wo = (const float*)d_in[10]; const float* bo = (const float*)d_in[11];

    float* out  = (float*)d_out;
    float* attn = out + OUT_ELEMS_;

    char* ws = (char*)d_ws;
    const size_t FSZ = (size_t)MTOT_ * D_ * 2;   // 8 MB per f16 matrix
    _Float16* Qh   = (_Float16*)(ws + 0 * FSZ);
    _Float16* Kh   = (_Float16*)(ws + 1 * FSZ);
    _Float16* Vh   = (_Float16*)(ws + 2 * FSZ);
    _Float16* Ctxh = (_Float16*)(ws + 3 * FSZ);

    dim3 ggrid(D_ / GBN, MTOT_ / GBM);           // (8, 32)
    gemm_bias_wmma<float, false><<<ggrid, 256, 0, stream>>>(queries, Wq, bq, Qh, MTOT_, D_, D_);
    gemm_bias_wmma<float, false><<<ggrid, 256, 0, stream>>>(keys,    Wk, bk, Kh, MTOT_, D_, D_);
    gemm_bias_wmma<float, false><<<ggrid, 256, 0, stream>>>(values,  Wv, bv, Vh, MTOT_, D_, D_);

    const int nblk = B_ * H_ * (S_ / QR);        // 4096
    attn_kernel<<<nblk, 256, ATTN_SMEM, stream>>>(Qh, Kh, Vh, mask, attn, Ctxh);

    gemm_bias_wmma<_Float16, true><<<ggrid, 256, 0, stream>>>(Ctxh, Wo, bo, out, MTOT_, D_, D_);
}
